// MultiHeadAttention_28132035789071
// MI455X (gfx1250) — compile-verified, matrix operand order checked by reading, not run on hardware
//
#include <hip/hip_runtime.h>
#include <hip/hip_bf16.h>
#include <math.h>

// ---------------------------------------------------------------------------
// Problem constants (B=1, N=M=512, Q=128, H=8, D=16, CLIP=10)
// ---------------------------------------------------------------------------
#define NN 512
#define MM 512
#define QQ 128
#define HH 8
#define DD 16
#define HD (HH * DD)   // 128
#define CLIPV 10.0f

// Padded LDS row strides (in elements). 128 bf16 rows padded by 8 bf16 so the
// per-lane stride is 272 B = 256+16 -> consecutive lanes land on different
// banks (64 banks x 4B period = 256 B).
#define ZSTR  136   // bf16 elements per z-tile row   (272 B)
#define WSTR  136   // bf16 elements per weight row   (272 B)
#define RYSTR 132   // f32 elements per r_y row       (528 B, 528%256=16)

typedef __attribute__((ext_vector_type(16))) __bf16    v16bf;
typedef __attribute__((ext_vector_type(8)))  float     v8f;
typedef __attribute__((ext_vector_type(4)))  unsigned  v4u;
typedef __attribute__((ext_vector_type(8)))  int       v8i;
typedef __attribute__((ext_vector_type(4)))  int       v4i;

// Workspace layout (bytes)
static constexpr size_t WS_WPACK = 0;                                            // 65536 B
static constexpr size_t WS_RX    = WS_WPACK + 2 * HH * DD * QQ * sizeof(unsigned short);
static constexpr size_t WS_RY    = WS_RX + (size_t)NN * HD * sizeof(float);
static constexpr size_t WS_XP    = WS_RY + (size_t)MM * HD * sizeof(float);
static constexpr size_t WS_R     = WS_XP + (size_t)NN * HD * sizeof(float);
static constexpr size_t WS_ATT   = WS_R  + (size_t)MM * HD * sizeof(float);      // 8 MB

// float -> bf16 (round to nearest even)
__device__ __forceinline__ unsigned short f2bf(float f) {
    unsigned int u = __float_as_uint(f);
    unsigned int r = u + 0x7fffu + ((u >> 16) & 1u);
    return (unsigned short)(r >> 16);
}

// K offset inside a 16-bit WMMA fragment register (ISA 7.12.2, 16-bit A 16x32):
// VGPR r holds K pair at (r<4?0:16) + (r&3)*2 + half*8
__device__ __forceinline__ int frag_k0(int r, int half) {
    return ((r & 4) << 2) + ((r & 3) << 1) + (half << 3);
}

// Load one v16bf fragment from an LDS row of 128 bf16 (padded stride applied
// by caller in 'rowbase'); per half this is two contiguous 16B runs -> the
// compiler merges to 2x ds_load_b128.
__device__ __forceinline__ v16bf load_frag(const unsigned short* rowbase, int kk, int half) {
    union { unsigned int u[8]; v16bf v; } f;
#pragma unroll
    for (int r = 0; r < 8; ++r) {
        int k0 = kk * 32 + frag_k0(r, half);
        f.u[r] = *(const unsigned int*)(rowbase + k0);
    }
    return f.v;
}

// ---------------------------------------------------------------------------
// TDM 1-D copy global->LDS with optional per-row LDS padding.
// Descriptor per ISA 8.3/8.4: group0 {count=1, lds_addr, global_addr, type=2},
// group1 {data_size=4B, pad ctrl, tensor_dim0=tile_dim0=ndw, dims1=1}.
// Issue from one wave only; fence with s_wait_tensorcnt.
// ---------------------------------------------------------------------------
#if __has_builtin(__builtin_amdgcn_tensor_load_to_lds)
#define HAVE_TDM 1
__device__ __forceinline__ void tdm_load_1d(void* lds_dst, const void* gsrc,
                                            unsigned ndw, int pad_int_code,
                                            int pad_amt_code, bool pad) {
    unsigned lds_off = (unsigned)(size_t)lds_dst;            // addr[31:0] = LDS offset
    unsigned long long ga = (unsigned long long)(size_t)gsrc;
    v4u g0 = {0u, 0u, 0u, 0u};
    g0[0] = 1u;                                              // count=1, user desc
    g0[1] = lds_off;
    g0[2] = (unsigned)(ga & 0xffffffffu);
    g0[3] = (unsigned)((ga >> 32) & 0x01ffffffu) | (2u << 30); // type=2 ("image")
    v8i g1 = {0, 0, 0, 0, 0, 0, 0, 0};
    unsigned dw0 = (2u << 16);                               // data_size = 4 B
    if (pad) dw0 |= (1u << 20) | ((unsigned)pad_int_code << 22)
                  | ((unsigned)pad_amt_code << 25);
    g1[0] = (int)dw0;
    g1[1] = (int)((ndw & 0xffffu) << 16);                    // tensor_dim0 lo16
    g1[2] = (int)(((ndw >> 16) & 0xffffu) | (1u << 16));     // dim0 hi16 | tensor_dim1=1
    g1[3] = (int)((ndw & 0xffffu) << 16);                    // tile_dim0 (<=65535)
    g1[4] = 1;                                               // tile_dim1 = 1
    g1[5] = (int)ndw;                                        // tensor_dim0_stride lo32
    v4i g2 = {0, 0, 0, 0}, g3 = {0, 0, 0, 0};
#if defined(__clang_major__) && (__clang_major__ >= 23)
    v8i g4 = {0, 0, 0, 0, 0, 0, 0, 0};
    __builtin_amdgcn_tensor_load_to_lds(g0, g1, g2, g3, g4, 0);
#else
    __builtin_amdgcn_tensor_load_to_lds(g0, g1, g2, g3, 0);
#endif
}
#else
#define HAVE_TDM 0
#endif

// ---------------------------------------------------------------------------
// Kernel 0a: pack lambda_z1/lambda_z2 [H][Q][D] f32 -> bf16 [s][h][d][q] (flat)
// ---------------------------------------------------------------------------
__global__ __launch_bounds__(256) void pack_w_kernel(
    const float* __restrict__ lz1, const float* __restrict__ lz2,
    unsigned short* __restrict__ wpack)
{
    int i = blockIdx.x * 256 + threadIdx.x;        // 32768 elements
    if (i >= 2 * HH * DD * QQ) return;
    int q = i & (QQ - 1);
    int d = (i >> 7) & (DD - 1);
    int h = (i >> 11) & (HH - 1);
    int s = i >> 14;
    const float* src = s ? lz2 : lz1;
    wpack[i] = f2bf(src[(h * QQ + q) * DD + d]);
}

// ---------------------------------------------------------------------------
// Kernel 0b: node projections r_x, r_y, x_p
// ---------------------------------------------------------------------------
__global__ __launch_bounds__(128) void node_proj_kernel(
    const float* __restrict__ x, const float* __restrict__ y,
    const float* __restrict__ lx, const float* __restrict__ ly,
    const float* __restrict__ bx, const float* __restrict__ by,
    const float* __restrict__ th1,
    float* __restrict__ r_x, float* __restrict__ r_y, float* __restrict__ x_p)
{
    int n  = blockIdx.x;
    int hd = threadIdx.x;
    int h  = hd >> 4, d = hd & 15;
    __shared__ float xs[QQ], ys[QQ];
    xs[hd] = x[n * QQ + hd];
    ys[hd] = y[n * QQ + hd];
    __syncthreads();
    float a1 = 0.f, a2 = 0.f, a3 = 0.f;
#pragma unroll 4
    for (int q = 0; q < QQ; ++q) {
        int wi = (h * QQ + q) * DD + d;
        a1 = fmaf(xs[q], lx[wi],  a1);
        a2 = fmaf(ys[q], ly[wi],  a2);
        a3 = fmaf(xs[q], th1[wi], a3);
    }
    r_x[n * HD + hd] = a1 + bx[hd];
    r_y[n * HD + hd] = a2 + by[hd];
    x_p[n * HD + hd] = a3;
}

// ---------------------------------------------------------------------------
// Score-tile epilogue: att[h][n][m0+row] = sum_d (rz1+rx)(rz2+ry), scale,
// 10*tanh, mask. C layout: col d = lane%16, row = vgpr + 8*(lane/16).
// ---------------------------------------------------------------------------
__device__ __forceinline__ void score_epilogue(
    v8f c1, v8f c2, int n_glob, int m0, int h, int half, int lm,
    const float* __restrict__ r_x, const float* ry_s,
    const int* __restrict__ mask, float* __restrict__ att)
{
    const float rxv = r_x[n_glob * HD + h * DD + lm];
    float prod[8];
#pragma unroll
    for (int r = 0; r < 8; ++r) {
        int row = r + 8 * half;
        float ryv = ry_s[row * RYSTR + h * DD + lm];
        prod[r] = (c1[r] + rxv) * (c2[r] + ryv);
    }
#pragma unroll
    for (int r = 0; r < 8; ++r) {
        float v = prod[r];
        v += __shfl_xor(v, 1, 32);
        v += __shfl_xor(v, 2, 32);
        v += __shfl_xor(v, 4, 32);
        v += __shfl_xor(v, 8, 32);
        prod[r] = v;
    }
    if (lm == 0) {      // lanes 0 / 16 write rows 0-7 / 8-15
#pragma unroll
        for (int r = 0; r < 8; ++r) {
            int m_glob = m0 + r + 8 * half;
            float v = prod[r] * 0.25f;           // 1/sqrt(D), D=16
            v = CLIPV * tanhf(v);
            if (mask[n_glob * MM + m_glob] == 1) v = -INFINITY;
            att[((size_t)h * NN + n_glob) * MM + m_glob] = v;
        }
    }
}

// ---------------------------------------------------------------------------
// Kernel 1: fused edge projection + attention scores (bf16 WMMA)
// grid=(M/16,N/16), block=256 (8 waves). Wave keeps A-frags for 2 n_locals
// resident; each B-frag pair feeds 4 WMMAs. Weights + r_y tile staged via TDM.
// ---------------------------------------------------------------------------
__global__ __launch_bounds__(256) void attn_scores_kernel(
    const float* __restrict__ z, const int* __restrict__ mask,
    const unsigned short* __restrict__ wpack,
    const float* __restrict__ r_x, const float* __restrict__ r_y,
    float* __restrict__ att)
{
    __shared__ unsigned short zt[16 * 16 * ZSTR];      // 69632 B: [n_l][m_l][q] (padded)
    __shared__ unsigned short wl[2 * HH * DD * WSTR];  // 69632 B: [s][h][d][q] (padded)
    __shared__ float          ry_s[16 * RYSTR];        //  8448 B: [m_l][h*16+d] (padded)

    const int m0  = blockIdx.x * 16;
    const int n0  = blockIdx.y * 16;
    const int tid = threadIdx.x;

    // ---- stage weights (64 KB) + r_y tile (8 KB) into padded LDS via TDM ----
#if HAVE_TDM
    if (tid < 32) {
        // 16384 dwords, pad 4 dwords (code 3) every 64 dwords (code 5)
        tdm_load_1d(wl, wpack, (2 * HH * DD * QQ) / 2, 5, 3, true);
        // 2048 dwords, pad 4 dwords every 128 dwords (code 6)
        tdm_load_1d(ry_s, r_y + (size_t)m0 * HD, 16 * HD, 6, 3, true);
        __builtin_amdgcn_s_wait_tensorcnt(0);
    }
#else
    {
        const unsigned int* src = (const unsigned int*)wpack;
        unsigned int* dst = (unsigned int*)wl;
#pragma unroll 4
        for (int i = tid; i < (2 * HH * DD * QQ) / 2; i += 256)
            dst[(i >> 6) * (WSTR / 2) + (i & 63)] = src[i];
#pragma unroll 2
        for (int i = tid; i < 16 * HD; i += 256)
            ry_s[(i >> 7) * RYSTR + (i & 127)] = r_y[(size_t)m0 * HD + i];
    }
#endif

    // ---- stage z tile (fp32 -> bf16), coalesced float4 reads ----
    {
        const float4* z4 = (const float4*)z;
#pragma unroll 4
        for (int i = tid; i < 16 * 16 * (QQ / 4); i += 256) {
            int p  = i >> 5;                 // pair index (n_l*16+m_l)
            int c4 = i & 31;                 // float4 index within q-row
            int n  = n0 + (p >> 4);
            int m  = m0 + (p & 15);
            float4 f = z4[(size_t)(n * MM + m) * (QQ / 4) + c4];
            unsigned int lo = (unsigned int)f2bf(f.x) | ((unsigned int)f2bf(f.y) << 16);
            unsigned int hi = (unsigned int)f2bf(f.z) | ((unsigned int)f2bf(f.w) << 16);
            unsigned int* drow = (unsigned int*)(zt + p * ZSTR + c4 * 4);
            drow[0] = lo; drow[1] = hi;
        }
    }
    __syncthreads();

    const int wave = tid >> 5;
    const int lane = tid & 31;
    const int half = lane >> 4;
    const int lm   = lane & 15;

    const int nlA = wave * 2, nlB = wave * 2 + 1;
    const unsigned short* arowA = zt + (nlA * 16 + lm) * ZSTR;
    const unsigned short* arowB = zt + (nlB * 16 + lm) * ZSTR;

    // A fragments for both n_locals, all 4 K-chunks (resident: 64 VGPRs)
    v16bf A0[4], A1[4];
#pragma unroll
    for (int kk = 0; kk < 4; ++kk) {
        A0[kk] = load_frag(arowA, kk, half);
        A1[kk] = load_frag(arowB, kk, half);
    }

#pragma unroll 1
    for (int h = 0; h < HH; ++h) {
        const unsigned short* b1row = wl + ((0 * HH + h) * DD + lm) * WSTR;
        const unsigned short* b2row = wl + ((1 * HH + h) * DD + lm) * WSTR;

        v8f a00, a01, a10, a11;
#pragma unroll
        for (int i = 0; i < 8; ++i) { a00[i] = 0.f; a01[i] = 0.f; a10[i] = 0.f; a11[i] = 0.f; }

        // double-buffered B fragments; each pair feeds 4 WMMAs
        v16bf b1 = load_frag(b1row, 0, half);
        v16bf b2 = load_frag(b2row, 0, half);
#pragma unroll
        for (int kk = 0; kk < 4; ++kk) {
            v16bf nb1 = b1, nb2 = b2;
            if (kk < 3) {
                nb1 = load_frag(b1row, kk + 1, half);
                nb2 = load_frag(b2row, kk + 1, half);
            }
            a00 = __builtin_amdgcn_wmma_f32_16x16x32_bf16(false, A0[kk], false, b1, (short)0, a00, false, false);
            a01 = __builtin_amdgcn_wmma_f32_16x16x32_bf16(false, A0[kk], false, b2, (short)0, a01, false, false);
            a10 = __builtin_amdgcn_wmma_f32_16x16x32_bf16(false, A1[kk], false, b1, (short)0, a10, false, false);
            a11 = __builtin_amdgcn_wmma_f32_16x16x32_bf16(false, A1[kk], false, b2, (short)0, a11, false, false);
            b1 = nb1; b2 = nb2;
        }

        score_epilogue(a00, a01, n0 + nlA, m0, h, half, lm, r_x, ry_s, mask, att);
        score_epilogue(a10, a11, n0 + nlB, m0, h, half, lm, r_x, ry_s, mask, att);
    }
}

// ---------------------------------------------------------------------------
// Kernel 2: softmax over n (axis -2). One thread per (h, m) column.
// ---------------------------------------------------------------------------
__global__ __launch_bounds__(256) void softmax_n_kernel(float* __restrict__ att)
{
    int h = blockIdx.x >> 1;
    int m = ((blockIdx.x & 1) << 8) + threadIdx.x;
    float* col = att + (size_t)h * NN * MM + m;
    float mx = -INFINITY;
    for (int n = 0; n < NN; ++n) mx = fmaxf(mx, col[(size_t)n * MM]);
    float s = 0.f;
    for (int n = 0; n < NN; ++n) s += __expf(col[(size_t)n * MM] - mx);
    float inv = 1.f / s;   // fully-masked column -> NaN, matching reference
    for (int n = 0; n < NN; ++n)
        col[(size_t)n * MM] = __expf(col[(size_t)n * MM] - mx) * inv;
}

// ---------------------------------------------------------------------------
// Kernel 3a: r[m][h][d] = sum_n w[h][n][m] * x_p[n][h][d] + bias_theta
// ---------------------------------------------------------------------------
__global__ __launch_bounds__(128) void weighted_sum_kernel(
    const float* __restrict__ att, const float* __restrict__ x_p,
    const float* __restrict__ bias_theta, float* __restrict__ rbuf)
{
    int m  = blockIdx.x;
    int hd = threadIdx.x;
    int h  = hd >> 4;
    const float* wcol = att + (size_t)h * NN * MM + m;
    float acc = 0.f;
#pragma unroll 4
    for (int n = 0; n < NN; ++n)
        acc = fmaf(wcol[(size_t)n * MM], x_p[n * HD + hd], acc);
    rbuf[m * HD + hd] = acc + bias_theta[hd];
}

// ---------------------------------------------------------------------------
// Kernel 3b: out[m][q] = sum_{h,d} r[m][h][d] * theta2[h][q][d]
// ---------------------------------------------------------------------------
__global__ __launch_bounds__(128) void out_proj_kernel(
    const float* __restrict__ rbuf, const float* __restrict__ th2,
    float* __restrict__ out)
{
    int m = blockIdx.x;
    int q = threadIdx.x;
    __shared__ float rs[HD];
    rs[q] = rbuf[m * HD + q];
    __syncthreads();
    float acc = 0.f;
#pragma unroll 4
    for (int hd = 0; hd < HD; ++hd) {
        int h = hd >> 4, d = hd & 15;
        acc = fmaf(rs[hd], th2[(h * QQ + q) * DD + d], acc);
    }
    out[m * QQ + q] = acc;
}

// ---------------------------------------------------------------------------
// Launch
// ---------------------------------------------------------------------------
extern "C" void kernel_launch(void* const* d_in, const int* in_sizes, int n_in,
                              void* d_out, int out_size, void* d_ws, size_t ws_size,
                              hipStream_t stream) {
    const float* x   = (const float*)d_in[0];
    const float* y   = (const float*)d_in[1];
    const float* z   = (const float*)d_in[2];
    const int*   mask= (const int*)  d_in[3];
    const float* lx  = (const float*)d_in[4];
    const float* ly  = (const float*)d_in[5];
    const float* lz1 = (const float*)d_in[6];
    const float* lz2 = (const float*)d_in[7];
    const float* blx = (const float*)d_in[8];
    const float* bly = (const float*)d_in[9];
    const float* th1 = (const float*)d_in[10];
    const float* th2 = (const float*)d_in[11];
    const float* bth = (const float*)d_in[12];
    float*       out = (float*)d_out;

    char* ws = (char*)d_ws;
    unsigned short* wpack = (unsigned short*)(ws + WS_WPACK);
    float* r_x  = (float*)(ws + WS_RX);
    float* r_y  = (float*)(ws + WS_RY);
    float* x_p  = (float*)(ws + WS_XP);
    float* rbuf = (float*)(ws + WS_R);
    float* att  = (float*)(ws + WS_ATT);

    pack_w_kernel<<<(2 * HH * DD * QQ + 255) / 256, 256, 0, stream>>>(lz1, lz2, wpack);
    node_proj_kernel<<<NN, HD, 0, stream>>>(x, y, lx, ly, blx, bly, th1, r_x, r_y, x_p);
    attn_scores_kernel<<<dim3(MM / 16, NN / 16), 256, 0, stream>>>(
        z, mask, wpack, r_x, r_y, att);
    softmax_n_kernel<<<HH * (MM / 256), 256, 0, stream>>>(att);
    weighted_sum_kernel<<<MM, HD, 0, stream>>>(att, x_p, bth, rbuf);
    out_proj_kernel<<<MM, HD, 0, stream>>>(rbuf, th2, out);
}
